// DimensionalFRR_58677843198036
// MI455X (gfx1250) — matrix operand order repaired and checked by reading, not compiled, not run on hardware
//
#include <hip/hip_runtime.h>
#include <hip/hip_bf16.h>

// Model dims
#define S_   512
#define D_   768
#define H_   12
#define HD_  64
#define DH_  4
#define CHD_ 192
#define V_   32000
#define L_   28   // N_SCALES * ITERS

#define LDW_ 40   // LDS row stride (elements): 80B rows -> 16B aligned for b128 async

typedef __attribute__((ext_vector_type(16))) __bf16 v16bf;
typedef __attribute__((ext_vector_type(8)))  float  v8f;

union FragBF { v16bf v; unsigned int u[8]; };

// per-lane async copy of 16 bytes global -> LDS, tracked by ASYNCcnt
__device__ __forceinline__ void async_g2l_b128(unsigned ldsoff, const void* gptr)
{
    asm volatile("global_load_async_to_lds_b128 %0, %1, off"
                 :: "v"(ldsoff), "v"((unsigned long long)gptr)
                 : "memory");
}
__device__ __forceinline__ void wait_asynccnt0()
{
    asm volatile("s_wait_asynccnt 0x0" ::: "memory");
}

// ---------------------------------------------------------------------------
// Generic bf16 x bf16 -> f32 WMMA GEMM.  C = A(MxK) * B(KxN)  (optional B^T).
// Block tile 128x128, K-step 32, 8 waves, each wave = 4x2 WMMA 16x16 tiles.
// A tile (and B when transB) filled via GLOBAL_LOAD_ASYNC_TO_LDS_B128 when the
// tile is fully in-bounds; non-trans B uses 16B vector loads + LDS scatter.
// Batched via grid.z with per-operand element strides. Output f32 or bf16.
// ---------------------------------------------------------------------------
__global__ __launch_bounds__(256) void k_gemm_bf16(
    const __bf16* __restrict__ A, long lda, long strideA,
    const __bf16* __restrict__ B, long ldb, long strideB,
    void* __restrict__ Cv, long ldc, long strideC,
    int M, int N, int K, int transB, int outBf)
{
    __shared__ __align__(16) __bf16 As[128][LDW_];   // [m][k]
    __shared__ __align__(16) __bf16 Bs[128][LDW_];   // [n][k], K contiguous per column

    const int bz = blockIdx.z;
    A += (long)bz * strideA;
    B += (long)bz * strideB;

    const int m0   = blockIdx.y * 128;
    const int n0   = blockIdx.x * 128;
    const int tid  = threadIdx.x;
    const int lane = tid & 31;
    const int wave = tid >> 5;
    const int wy   = wave >> 2;   // 0..1 -> 64 rows each
    const int wx   = wave & 3;    // 0..3 -> 32 cols each

    v8f acc[4][2] = {};

    // ISA lane layouts (16-bit operands, wave32):
    // A 16x32: lanes 0-15 row M=lane hold K pairs {0..7,16..23}; lanes 16-31 hold {8..15,24..31}
    // B 32x16: lanes 0-15 col N=lane hold K pairs {0..7,16..23}; lanes 16-31 hold {8..15,24..31}
    const int arow = wy * 64 + (lane & 15);
    const int kb   = (lane < 16) ? 0 : 8;
    const int brow = wx * 32 + (lane & 15);
    const int kh   = (lane < 16) ? 0 : 16;

    const bool aFullM = (m0 + 128 <= M);
    const bool bFullN = (n0 + 128 <= N);

    for (int k0 = 0; k0 < K; k0 += 32) {
        const bool kFull = (k0 + 32 <= K);

        // ---- A tile: 128 x 32 ----
        if (aFullM && kFull) {
            // async DMA: 512 chunks of 16B, 2 per thread, no VGPR roundtrip
            #pragma unroll
            for (int c = tid; c < 512; c += 256) {
                int r  = c >> 2;
                int ko = (c & 3) * 8;
                async_g2l_b128((unsigned)(unsigned long long)(const void*)&As[r][ko],
                               A + (long)(m0 + r) * lda + (k0 + ko));
            }
        } else {
            for (int e = tid; e < 128 * 32; e += 256) {
                int r = e >> 5, c = e & 31;
                int gr = m0 + r, gc = k0 + c;
                __bf16 va = (__bf16)0.0f;
                if (gr < M && gc < K) va = A[(long)gr * lda + gc];
                As[r][c] = va;
            }
        }

        // ---- B tile -> Bs[n][k] ----
        if (transB) {
            if (bFullN && kFull) {
                #pragma unroll
                for (int c = tid; c < 512; c += 256) {
                    int n  = c >> 2;
                    int ko = (c & 3) * 8;
                    async_g2l_b128((unsigned)(unsigned long long)(const void*)&Bs[n][ko],
                                   B + (long)(n0 + n) * ldb + (k0 + ko));
                }
            } else {
                for (int e = tid; e < 128 * 32; e += 256) {
                    int n = e >> 5, c = e & 31;
                    int gn = n0 + n, gc = k0 + c;
                    __bf16 vb = (__bf16)0.0f;
                    if (gn < N && gc < K) vb = B[(long)gn * ldb + gc];
                    Bs[n][c] = vb;
                }
            }
        } else {
            // global is N-contiguous: 16B vector load (8 cols) + 8-row LDS scatter
            #pragma unroll
            for (int c = tid; c < 512; c += 256) {
                int kk = c >> 4;            // 0..31
                int nn = (c & 15) * 8;      // 0..120
                int gk = k0 + kk;
                const __bf16* src = B + (long)gk * ldb + (n0 + nn);
                __bf16 tmp[8];
                if (gk < K && (n0 + nn + 8) <= N) {
                    float4 v = *(const float4*)src;
                    const __bf16* pv = (const __bf16*)&v;
                    #pragma unroll
                    for (int e = 0; e < 8; e++) tmp[e] = pv[e];
                } else {
                    #pragma unroll
                    for (int e = 0; e < 8; e++) {
                        int gn = n0 + nn + e;
                        tmp[e] = (gk < K && gn < N) ? src[e] : (__bf16)0.0f;
                    }
                }
                #pragma unroll
                for (int e = 0; e < 8; e++) Bs[nn + e][kk] = tmp[e];
            }
        }

        wait_asynccnt0();     // drain async global->LDS copies from this wave
        __syncthreads();

        FragBF a[4], b[2];
        #pragma unroll
        for (int i = 0; i < 4; i++) {
            const __bf16* rp = &As[arow + i * 16][0];
            #pragma unroll
            for (int t = 0; t < 8; t++) {
                int k = ((t >> 2) << 4) + kb + ((t & 3) << 1);   // K pair base
                a[i].u[t] = *(const unsigned int*)(rp + k);
            }
        }
        #pragma unroll
        for (int j = 0; j < 2; j++) {
            const __bf16* rp = &Bs[brow + j * 16][0];
            #pragma unroll
            for (int t = 0; t < 8; t++)
                b[j].u[t] = *(const unsigned int*)(rp + ((t >> 2) << 4) + kb + ((t & 3) << 1));
        }

        #pragma unroll
        for (int i = 0; i < 4; i++)
            #pragma unroll
            for (int j = 0; j < 2; j++)
                acc[i][j] = __builtin_amdgcn_wmma_f32_16x16x32_bf16(
                    false, a[i].v, false, b[j].v, (short)0, acc[i][j], false, false);
        __syncthreads();
    }

    // C/D layout: VGPR r -> row r (lanes 0-15) / row r+8 (lanes 16-31), col = lane&15
    const int cn   = lane & 15;
    const int rofs = (lane < 16) ? 0 : 8;
    float*  Cf = (float*)Cv  + (long)bz * strideC;
    __bf16* Cb = (__bf16*)Cv + (long)bz * strideC;
    for (int i = 0; i < 4; i++)
        for (int j = 0; j < 2; j++) {
            int col = n0 + wx * 32 + j * 16 + cn;
            if (col >= N) continue;
            #pragma unroll
            for (int r = 0; r < 8; r++) {
                int row = m0 + wy * 64 + i * 16 + rofs + r;
                if (row < M) {
                    float v = acc[i][j][r];
                    if (outBf) Cb[(long)row * ldc + col] = (__bf16)v;
                    else       Cf[(long)row * ldc + col] = v;
                }
            }
        }
}

// ---------------------------------------------------------------------------
// Elementwise / reduction helpers
// ---------------------------------------------------------------------------
__global__ void k_cvt_bf16(const float* __restrict__ in, __bf16* __restrict__ out, long n)
{
    long i = (long)blockIdx.x * 256 + threadIdx.x;
    if (i < n) out[i] = (__bf16)in[i];
}

__global__ void k_embed(const int* __restrict__ tok, const float* __restrict__ emb,
                        float* __restrict__ x)
{
    int s = blockIdx.x;
    long base = (long)tok[s] * D_;
    for (int d = threadIdx.x; d < D_; d += 256)
        x[(long)s * D_ + d] = emb[base + d];
}

// h = rms(x) * gamma + beta ; also emits bf16 copy.  One block per row.
__global__ void k_rms_mod(const float* __restrict__ x, const float* __restrict__ gamma,
                          const float* __restrict__ beta, float* __restrict__ h,
                          __bf16* __restrict__ hbf)
{
    int s = blockIdx.x;
    const float* row = x + (size_t)s * D_;
    __shared__ float red[8];
    float ss = 0.f;
    for (int d = threadIdx.x; d < D_; d += 256) { float v = row[d]; ss += v * v; }
    for (int o = 16; o > 0; o >>= 1) ss += __shfl_down(ss, o, 32);
    if ((threadIdx.x & 31) == 0) red[threadIdx.x >> 5] = ss;
    __syncthreads();
    if (threadIdx.x == 0) {
        float t = 0.f;
        for (int i = 0; i < 8; i++) t += red[i];
        red[0] = rsqrtf(t / (float)D_ + 1e-6f);
    }
    __syncthreads();
    float r = red[0];
    for (int d = threadIdx.x; d < D_; d += 256) {
        float v = row[d] * r * gamma[d] + (beta ? beta[d] : 0.f);
        h[(size_t)s * D_ + d]   = v;
        hbf[(size_t)s * D_ + d] = (__bf16)v;
    }
}

// causal softmax over scores (H,S,S) f32 -> probs bf16; applies 1/sqrt(64)
__global__ void k_softmax_causal(const float* __restrict__ sc, __bf16* __restrict__ p)
{
    int q = blockIdx.x, h = blockIdx.y;
    const float* row  = sc + ((long)h * S_ + q) * (long)S_;
    __bf16*      prow = p  + ((long)h * S_ + q) * (long)S_;
    __shared__ float red[8];
    int tid = threadIdx.x;
    float mx = -3.4e38f;
    for (int k = tid; k < S_; k += 256)
        if (k <= q) mx = fmaxf(mx, row[k] * 0.125f);
    for (int o = 16; o > 0; o >>= 1) mx = fmaxf(mx, __shfl_down(mx, o, 32));
    if ((tid & 31) == 0) red[tid >> 5] = mx;
    __syncthreads();
    if (tid == 0) {
        float m = red[0];
        for (int i = 1; i < 8; i++) m = fmaxf(m, red[i]);
        red[0] = m;
    }
    __syncthreads();
    mx = red[0];
    __syncthreads();
    float sum = 0.f;
    for (int k = tid; k < S_; k += 256)
        if (k <= q) sum += __expf(row[k] * 0.125f - mx);
    for (int o = 16; o > 0; o >>= 1) sum += __shfl_down(sum, o, 32);
    if ((tid & 31) == 0) red[tid >> 5] = sum;
    __syncthreads();
    if (tid == 0) {
        float t = 0.f;
        for (int i = 0; i < 8; i++) t += red[i];
        red[0] = t;
    }
    __syncthreads();
    float inv = 1.f / red[0];
    for (int k = tid; k < S_; k += 256) {
        float v = (k <= q) ? __expf(row[k] * 0.125f - mx) * inv : 0.f;
        prow[k] = (__bf16)v;
    }
}

__global__ void k_add(const float* __restrict__ a, const float* __restrict__ b,
                      float* __restrict__ o, int n)
{
    int i = blockIdx.x * 256 + threadIdx.x;
    if (i < n) o[i] = a[i] + b[i];
}

// gelu (tanh approx, jax default) with bf16 output
__global__ void k_gelu_cvt(const float* __restrict__ in, __bf16* __restrict__ out, int n)
{
    int i = blockIdx.x * 256 + threadIdx.x;
    if (i < n) {
        float x = in[i];
        float c = 0.7978845608028654f * (x + 0.044715f * x * x * x);
        out[i] = (__bf16)(0.5f * x * (1.f + tanhf(c)));
    }
}

// xb = x + (x1 + f - x) * iter_scale[idx] ; also bf16 copy
__global__ void k_iterblend(const float* __restrict__ x, const float* __restrict__ x1,
                            const float* __restrict__ f, const float* __restrict__ is,
                            int idx, float* __restrict__ xb, __bf16* __restrict__ xbbf, int n)
{
    int i = blockIdx.x * 256 + threadIdx.x;
    if (i < n) {
        float s = is[idx];
        float v = x[i] + (x1[i] + f[i] - x[i]) * s;
        xb[i] = v;  xbbf[i] = (__bf16)v;
    }
}

// x = (1-g)*xb + g*(xb + ocd) = xb + g*ocd, g = sigmoid(depth_gate[idx])
__global__ void k_gate(const float* __restrict__ xb, const float* __restrict__ ocd,
                       const float* __restrict__ dg, int idx,
                       float* __restrict__ xg, __bf16* __restrict__ xgbf, int n)
{
    int i = blockIdx.x * 256 + threadIdx.x;
    if (i < n) {
        float g = 1.f / (1.f + __expf(-dg[idx]));
        float v = xb[i] + g * ocd[i];
        xg[i] = v;  xgbf[i] = (__bf16)v;
    }
}

// cross-depth attention over cached K/V history: grid (S, DH), 64 threads/block
__global__ void k_cross_attn(const float* __restrict__ qc, const __bf16* __restrict__ Kh,
                             const __bf16* __restrict__ Vh, __bf16* __restrict__ od, int nhist)
{
    int s = blockIdx.x, h = blockIdx.y;
    __shared__ float sc[32];
    __shared__ float red[2];
    const float* qp = qc + (long)s * D_ + h * CHD_;

    for (int i = 0; i < nhist; i++) {
        const __bf16* kp = Kh + ((long)i * S_ + s) * D_ + h * CHD_;
        float d = 0.f;
        for (int e = threadIdx.x; e < CHD_; e += 64) d += qp[e] * (float)kp[e];
        for (int o = 16; o > 0; o >>= 1) d += __shfl_down(d, o, 32);
        if ((threadIdx.x & 31) == 0) red[threadIdx.x >> 5] = d;
        __syncthreads();
        if (threadIdx.x == 0) sc[i] = (red[0] + red[1]) * 0.07216878364870323f; // 1/sqrt(192)
        __syncthreads();
    }
    if (threadIdx.x == 0) {
        float m = -3.4e38f;
        for (int i = 0; i < nhist; i++) m = fmaxf(m, sc[i]);
        float t = 0.f;
        for (int i = 0; i < nhist; i++) { float e = __expf(sc[i] - m); sc[i] = e; t += e; }
        red[0] = 1.f / t;
    }
    __syncthreads();
    float inv = red[0];
    for (int e = threadIdx.x; e < CHD_; e += 64) {
        float acc = 0.f;
        for (int i = 0; i < nhist; i++)
            acc += sc[i] * (float)Vh[((long)i * S_ + s) * D_ + h * CHD_ + e];
        od[(long)s * D_ + h * CHD_ + e] = (__bf16)(acc * inv);
    }
}

// ---------------------------------------------------------------------------
extern "C" void kernel_launch(void* const* d_in, const int* in_sizes, int n_in,
                              void* d_out, int out_size, void* d_ws, size_t ws_size,
                              hipStream_t stream)
{
    const int*   tokens = (const int*)  d_in[0];
    const float* embed  = (const float*)d_in[1];
    const float* Wsrc[10] = {           // wq wk wv wo w1 w2 cq ck cv co
        (const float*)d_in[2], (const float*)d_in[3], (const float*)d_in[4],
        (const float*)d_in[5], (const float*)d_in[6], (const float*)d_in[7],
        (const float*)d_in[8], (const float*)d_in[9], (const float*)d_in[10],
        (const float*)d_in[11] };
    const float* scale_gamma = (const float*)d_in[12];
    const float* scale_beta  = (const float*)d_in[13];
    const float* iter_scale  = (const float*)d_in[14];
    const float* depth_gate  = (const float*)d_in[15];
    const float* norm_w      = (const float*)d_in[16];
    const float* lm_head     = (const float*)d_in[17];
    float* out = (float*)d_out;
    (void)in_sizes; (void)n_in; (void)out_size; (void)ws_size;

    char* pool = (char*)d_ws;
    size_t off = 0;
    auto alloc = [&](size_t bytes) -> void* {
        void* r = pool + off;
        off = (off + bytes + 255) & ~(size_t)255;
        return r;
    };
    const size_t SD = (size_t)S_ * D_;

    // workspace
    __bf16* Wbf[10];
    for (int i = 0; i < 10; i++) Wbf[i] = (__bf16*)alloc((size_t)D_ * D_ * 2);
    __bf16* lmbf = (__bf16*)alloc((size_t)D_ * V_ * 2);
    __bf16* Kh   = (__bf16*)alloc((size_t)L_ * SD * 2);
    __bf16* Vh   = (__bf16*)alloc((size_t)L_ * SD * 2);
    float*  scores = (float*)alloc((size_t)H_ * S_ * S_ * 4);
    __bf16* pbf    = (__bf16*)alloc((size_t)H_ * S_ * S_ * 2);
    float* rot[3];
    for (int i = 0; i < 3; i++) rot[i] = (float*)alloc(SD * 4);
    float* hbuf  = (float*)alloc(SD * 4);
    float* oproj = (float*)alloc(SD * 4);
    float* x1    = (float*)alloc(SD * 4);
    float* t1    = (float*)alloc(SD * 4);
    float* f2    = (float*)alloc(SD * 4);
    float* qc    = (float*)alloc(SD * 4);
    float* ocd   = (float*)alloc(SD * 4);
    __bf16* hbf    = (__bf16*)alloc(SD * 2);
    __bf16* qbf    = (__bf16*)alloc(SD * 2);
    __bf16* kbf    = (__bf16*)alloc(SD * 2);
    __bf16* vbf    = (__bf16*)alloc(SD * 2);
    __bf16* attnbf = (__bf16*)alloc(SD * 2);
    __bf16* t1bf   = (__bf16*)alloc(SD * 2);
    __bf16* odbf   = (__bf16*)alloc(SD * 2);
    __bf16* xbbf   = (__bf16*)alloc(SD * 2);
    __bf16* xgbf   = (__bf16*)alloc(SD * 2);
    __bf16* ybf    = (__bf16*)alloc(SD * 2);

    auto gemm = [&](const __bf16* A, long lda, long sA,
                    const __bf16* B, long ldb, long sB,
                    void* C, long ldc, long sC,
                    int M, int N, int K, int transB, int outBf, int batch) {
        dim3 grid((unsigned)((N + 127) / 128), (unsigned)((M + 127) / 128), (unsigned)batch);
        k_gemm_bf16<<<grid, 256, 0, stream>>>(A, lda, sA, B, ldb, sB, C, ldc, sC,
                                              M, N, K, transB, outBf);
    };

    // --- weight conversion to bf16 (deterministic each call) ---
    {
        long n = (long)D_ * D_;
        dim3 g((unsigned)((n + 255) / 256));
        for (int i = 0; i < 10; i++)
            k_cvt_bf16<<<g, 256, 0, stream>>>(Wsrc[i], Wbf[i], n);
        long nl = (long)D_ * V_;
        k_cvt_bf16<<<dim3((unsigned)((nl + 255) / 256)), 256, 0, stream>>>(lm_head, lmbf, nl);
    }

    // --- embedding ---
    k_embed<<<dim3(S_), 256, 0, stream>>>(tokens, embed, rot[0]);
    int ix = 0;

    const int nSD = (int)SD;
    dim3 gE((unsigned)((nSD + 255) / 256));

    for (int l = 0; l < L_; ++l) {
        int sc = l / 7;
        int ib = (ix + 1) % 3, ig = (ix + 2) % 3;
        const float* gamma = scale_gamma + (size_t)sc * D_;
        const float* beta  = scale_beta  + (size_t)sc * D_;
        float* x = rot[ix];

        // --- self-attention ---
        k_rms_mod<<<dim3(S_), 256, 0, stream>>>(x, gamma, beta, hbuf, hbf);
        gemm(hbf, D_, 0, Wbf[0], D_, 0, qbf, D_, 0, S_, D_, D_, 0, 1, 1);
        gemm(hbf, D_, 0, Wbf[1], D_, 0, kbf, D_, 0, S_, D_, D_, 0, 1, 1);
        gemm(hbf, D_, 0, Wbf[2], D_, 0, vbf, D_, 0, S_, D_, D_, 0, 1, 1);
        // scores[h] = Q_h (512x64) * K_h^T  (batched over heads, async LDS fill both sides)
        gemm(qbf, D_, HD_, kbf, D_, HD_, scores, S_, (long)S_ * S_, S_, S_, HD_, 1, 0, H_);
        k_softmax_causal<<<dim3(S_, H_), 256, 0, stream>>>(scores, pbf);
        // attn[h] = P_h (512x512) * V_h (512x64) -> bf16 directly
        gemm(pbf, S_, (long)S_ * S_, vbf, D_, HD_, attnbf, D_, HD_, S_, HD_, S_, 0, 1, H_);
        gemm(attnbf, D_, 0, Wbf[3], D_, 0, oproj, D_, 0, S_, D_, D_, 0, 0, 1);
        k_add<<<gE, 256, 0, stream>>>(x, oproj, x1, nSD);

        // --- FFN ---
        k_rms_mod<<<dim3(S_), 256, 0, stream>>>(x1, gamma, beta, hbuf, hbf);
        gemm(hbf, D_, 0, Wbf[4], D_, 0, t1, D_, 0, S_, D_, D_, 0, 0, 1);
        k_gelu_cvt<<<gE, 256, 0, stream>>>(t1, t1bf, nSD);
        gemm(t1bf, D_, 0, Wbf[5], D_, 0, f2, D_, 0, S_, D_, D_, 0, 0, 1);

        // xb = x + (xblk - x) * iter_scale[l]
        k_iterblend<<<gE, 256, 0, stream>>>(x, x1, f2, iter_scale, l, rot[ib], xbbf, nSD);

        const __bf16* finbf;
        if (l == 0) {
            finbf = xbbf;  ix = ib;
        } else {
            // cross-depth attention over cached K/V history (KV computed once per entry)
            gemm(xbbf, D_, 0, Wbf[6], D_, 0, qc, D_, 0, S_, D_, D_, 0, 0, 1);
            k_cross_attn<<<dim3(S_, DH_), 64, 0, stream>>>(qc, Kh, Vh, odbf, l);
            gemm(odbf, D_, 0, Wbf[9], D_, 0, ocd, D_, 0, S_, D_, D_, 0, 0, 1);
            k_gate<<<gE, 256, 0, stream>>>(rot[ib], ocd, depth_gate, l, rot[ig], xgbf, nSD);
            finbf = xgbf;  ix = ig;
        }
        // append to history: K_l = x @ ck, V_l = x @ cv  (bf16, cached forever)
        gemm(finbf, D_, 0, Wbf[7], D_, 0, Kh + (size_t)l * SD, D_, 0, S_, D_, D_, 0, 1, 1);
        gemm(finbf, D_, 0, Wbf[8], D_, 0, Vh + (size_t)l * SD, D_, 0, S_, D_, D_, 0, 1, 1);
    }

    // --- final norm + LM head ---
    k_rms_mod<<<dim3(S_), 256, 0, stream>>>(rot[ix], norm_w, nullptr, hbuf, ybf);
    gemm(ybf, D_, 0, lmbf, V_, 0, out, V_, 0, S_, V_, D_, 0, 0, 1);
}
